// MambaBlock_23304492548273
// MI455X (gfx1250) — matrix operand order repaired and checked by reading, not compiled, not run on hardware
//
#include <hip/hip_runtime.h>
#include <hip/hip_bf16.h>
#include <math.h>

// ---- problem constants (reference: B=4, T=2048, D=1024, N=1024) ----
#define B_SZ   4
#define T_DIM  2048
#define D_DIM  1024
#define N_DIM  1024
#define KDIM   1024
#define M_TOT  (B_SZ * T_DIM)      // 8192 rows of the two GEMMs
#define LDT    40                  // padded LDS row stride (32 K + 8 pad) in bf16 elems

typedef __attribute__((ext_vector_type(16))) __bf16 v16bf;
typedef __attribute__((ext_vector_type(8)))  float  v8f;

union FragU { v16bf v; uint4 q[2]; };

__device__ __forceinline__ __bf16 f2bf(float f) {
    unsigned u = __builtin_bit_cast(unsigned, f);
    unsigned r = (u + 0x7FFFu + ((u >> 16) & 1u)) >> 16;   // round-nearest-even
    unsigned short s = (unsigned short)r;
    return __builtin_bit_cast(__bf16, s);
}

// CDNA5 async global->LDS copy (ASYNCcnt path, ISA 08_async_tensor §4).
// INST_OFFSET is added to BOTH the LDS and the global address, and our LDS
// destinations are byte-contiguous exactly like the global sources, so the
// sub-chunk offsets live in the instruction, not in address VGPRs.
#define ASYNC_CP_B128(ldsAddr, gaddr, OFFSTR)                                  \
    asm volatile("global_load_async_to_lds_b128 %0, %1, off offset:" OFFSTR    \
                 :: "v"(ldsAddr), "v"((unsigned long long)(gaddr))             \
                 : "memory")

__device__ __forceinline__ void wait_asynccnt0() {
    asm volatile("s_wait_asynccnt 0x0" ::: "memory");
}
__device__ __forceinline__ unsigned lds_addr_of(const void* p) {
    return (unsigned)(unsigned long long)p;   // generic->LDS: low 32 bits are the LDS offset
}

// ---------------- fp32 -> bf16 convert ----------------
__global__ void cvt_bf16_kernel(const float* __restrict__ src,
                                __bf16* __restrict__ dst, int n) {
    int i = blockIdx.x * blockDim.x + threadIdx.x;
    if (i < n) dst[i] = f2bf(src[i]);
}

// ---------------- delta = clip(softplus(x . delta_proj)) ----------------
__global__ void delta_kernel(const float* __restrict__ x,
                             const float* __restrict__ dproj,
                             float* __restrict__ delta) {
    int wave = threadIdx.x >> 5;
    int lane = threadIdx.x & 31;
    int row  = blockIdx.x * 8 + wave;
    const float* xr = x + (size_t)row * D_DIM;
    float s = 0.f;
    for (int i = lane; i < D_DIM; i += 32) s += xr[i] * dproj[i];
    for (int off = 16; off > 0; off >>= 1) s += __shfl_down(s, off, 32);
    if (lane == 0) {
        float sp = (s > 20.f) ? s : log1pf(__expf(s));
        delta[row] = fminf(fmaxf(sp, 0.001f), 0.1f);
    }
}

// ---------------- bf16 WMMA GEMM: C[m,n] = sum_k A[m,k]*Bm[n,k] ----------------
// block tile 128(M) x 256(N); 8 waves, each owns a 64x64 sub-tile (acc[4][4]).
// MODE 0: epilogue = silu + discretization -> BuF (fp32)
// MODE 1: epilogue = plain store -> outF (fp32)
template <int MODE>
__global__ __launch_bounds__(256) void gemm_bf16_kernel(
    const __bf16* __restrict__ A,    // [M_TOT, KDIM] row-major
    const __bf16* __restrict__ Bm,   // [N_DIM, KDIM] row-major (output-col major)
    float* __restrict__ outF,        // MODE 1 destination
    float* __restrict__ BuF,         // MODE 0 destination
    const float* __restrict__ delta,
    const float* __restrict__ A_log,
    const float* __restrict__ B_mat)
{
    __shared__ __bf16 Atile[128 * LDT];   // 10240 B
    __shared__ __bf16 Btile[256 * LDT];   // 20480 B

    const int m0 = blockIdx.y * 128;
    const int n0 = blockIdx.x * 256;

    const int t  = threadIdx.x;
    const int ar = t >> 1;           // A stage: row 0..127
    const int ap = t & 1;            // A stage: which 32B half

    const int lane   = t & 31;
    const int waveId = t >> 5;       // 8 waves
    const int wave_m = waveId & 1;   // 2 waves along M (x64)
    const int wave_n = waveId >> 1;  // 4 waves along N (x64)
    const int mloc   = lane & 15;
    const int khi    = lane >> 4;

    const __bf16* aSrc = A  + (size_t)(m0 + ar) * KDIM + ap * 16;
    const __bf16* bSrc = Bm + (size_t)(n0 + t) * KDIM;

    const unsigned aDst = lds_addr_of(Atile + ar * LDT + ap * 16);
    const unsigned bDst = lds_addr_of(Btile + t * LDT);

    v8f acc[4][4];
    const v8f vzero = {0.f, 0.f, 0.f, 0.f, 0.f, 0.f, 0.f, 0.f};
    for (int mi = 0; mi < 4; ++mi)
        for (int ni = 0; ni < 4; ++ni) acc[mi][ni] = vzero;

#pragma unroll 1
    for (int k0 = 0; k0 < KDIM; k0 += 32) {
        // ---- async stage: A slab 128x32 (32B/thread), B slab 256x32 (64B/thread) ----
        ASYNC_CP_B128(aDst, aSrc + k0, "0");
        ASYNC_CP_B128(aDst, aSrc + k0, "16");
        ASYNC_CP_B128(bDst, bSrc + k0, "0");
        ASYNC_CP_B128(bDst, bSrc + k0, "16");
        ASYNC_CP_B128(bDst, bSrc + k0, "32");
        ASYNC_CP_B128(bDst, bSrc + k0, "48");
        wait_asynccnt0();
        __syncthreads();

        // ---- A fragments (ISA 16-bit A layout), loaded once, reused 4x ----
        FragU afr[4];
#pragma unroll
        for (int mi = 0; mi < 4; ++mi) {
            const __bf16* p = Atile + (wave_m * 64 + mi * 16 + mloc) * LDT;
            afr[mi].q[0] = *(const uint4*)(p + khi * 8);
            afr[mi].q[1] = *(const uint4*)(p + 16 + khi * 8);
        }
        // ---- B fragments streamed; 16 WMMAs per K-step per wave ----
#pragma unroll
        for (int ni = 0; ni < 4; ++ni) {
            const __bf16* p = Btile + (wave_n * 64 + ni * 16 + mloc) * LDT + khi * 8;
            FragU bfr;
            bfr.q[0] = *(const uint4*)(p);
            bfr.q[1] = *(const uint4*)(p + 16);
#pragma unroll
            for (int mi = 0; mi < 4; ++mi)
                acc[mi][ni] = __builtin_amdgcn_wmma_f32_16x16x32_bf16(
                    false, afr[mi].v, false, bfr.v,
                    (short)0, acc[mi][ni], false, false);
        }
        __syncthreads();
    }

    // ---- epilogue: C/D layout row=(khi*8+r), col=mloc; fully unrolled so the
    // accumulator array is never dynamically indexed (no scratch round-trip) ----
#pragma unroll
    for (int ni = 0; ni < 4; ++ni) {
        const int gcol = n0 + wave_n * 64 + ni * 16 + mloc;
        float abar = 0.f, scale = 0.f;
        if (MODE == 0) {
            abar = __expf(A_log[gcol]);
            float dn = fmaxf(fabsf(abar), 1e-8f);
            scale = B_mat[gcol] * __builtin_amdgcn_rcpf(dn);   // v_rcp_f32, no div chain
        }
#pragma unroll
        for (int mi = 0; mi < 4; ++mi) {
            const int rbase = m0 + wave_m * 64 + mi * 16 + khi * 8;
#pragma unroll
            for (int r = 0; r < 8; ++r) {
                const int grow = rbase + r;
                float v = acc[mi][ni][r];
                if (MODE == 0) {
                    // silu via hardware reciprocal (bf16-input pipeline; 1-ulp rcp is fine)
                    float u  = v * __builtin_amdgcn_rcpf(1.f + __expf(-v));
                    float At = __expf(delta[grow] * abar);
                    BuF[(size_t)grow * N_DIM + gcol] = (At - 1.f) * scale * u;
                } else {
                    outF[(size_t)grow * N_DIM + gcol] = v;
                }
            }
        }
    }
}

// ---------------- sequential scan over T per (b,n) channel ----------------
__global__ void scan_kernel(const float* __restrict__ Bu,
                            const float* __restrict__ delta,
                            const float* __restrict__ A_log,
                            __bf16* __restrict__ hsbf)
{
    int idx = blockIdx.x * blockDim.x + threadIdx.x;   // 0..B*N-1
    int b   = idx >> 10;
    int nn  = idx & (N_DIM - 1);
    float abar = __expf(A_log[nn]);
    float h = 0.f;
    const float* dl = delta + (size_t)b * T_DIM;
    const float* bu = Bu   + (size_t)b * T_DIM * N_DIM + nn;
    __bf16*     out = hsbf + (size_t)b * T_DIM * N_DIM + nn;
    for (int tt = 0; tt < T_DIM; ++tt) {
        float a = __expf(dl[tt] * abar);
        h = __builtin_fmaf(a, h, bu[(size_t)tt * N_DIM]);
        out[(size_t)tt * N_DIM] = f2bf(h);
    }
}

// ---------------- host launcher ----------------
extern "C" void kernel_launch(void* const* d_in, const int* in_sizes, int n_in,
                              void* d_out, int out_size, void* d_ws, size_t ws_size,
                              hipStream_t stream)
{
    const float* x     = (const float*)d_in[0];   // [B,T,D]
    const float* W_in  = (const float*)d_in[1];   // [N,D]
    const float* W_out = (const float*)d_in[2];   // [D,N]
    const float* A_log = (const float*)d_in[3];   // [N]
    const float* B_mat = (const float*)d_in[4];   // [N]
    const float* dproj = (const float*)d_in[5];   // [D]
    float* y = (float*)d_out;                     // [B,T,D] fp32

    char* p = (char*)d_ws;
    __bf16* xbf    = (__bf16*)p; p += (size_t)M_TOT * D_DIM * 2;   // 16 MB
    __bf16* winbf  = (__bf16*)p; p += (size_t)N_DIM * D_DIM * 2;   //  2 MB
    __bf16* woutbf = (__bf16*)p; p += (size_t)D_DIM * N_DIM * 2;   //  2 MB
    float*  delta  = (float*)p;  p += (size_t)M_TOT * 4;           // 32 KB
    float*  Bu     = (float*)p;  p += (size_t)M_TOT * N_DIM * 4;   // 32 MB
    __bf16* hsbf   = (__bf16*)p;                                   // 16 MB

    const int nx = M_TOT * D_DIM;   // 8388608
    const int nw = N_DIM * D_DIM;   // 1048576
    cvt_bf16_kernel<<<nx / 256, 256, 0, stream>>>(x, xbf, nx);
    cvt_bf16_kernel<<<nw / 256, 256, 0, stream>>>(W_in, winbf, nw);
    cvt_bf16_kernel<<<nw / 256, 256, 0, stream>>>(W_out, woutbf, nw);

    delta_kernel<<<M_TOT / 8, 256, 0, stream>>>(x, dproj, delta);

    dim3 ggrid(N_DIM / 256, M_TOT / 128);   // 4 x 64 tiles
    gemm_bf16_kernel<0><<<ggrid, 256, 0, stream>>>(xbf, winbf, nullptr, Bu,
                                                   delta, A_log, B_mat);

    scan_kernel<<<(B_SZ * N_DIM) / 256, 256, 0, stream>>>(Bu, delta, A_log, hsbf);

    gemm_bf16_kernel<1><<<ggrid, 256, 0, stream>>>(hsbf, woutbf, y, nullptr,
                                                   nullptr, nullptr, nullptr);
}
